// TrajectoryEncoder_893353197714
// MI455X (gfx1250) — matrix operand order repaired and checked by reading, not compiled, not run on hardware
//
#include <hip/hip_runtime.h>
#include <hip/hip_bf16.h>
#include <stdint.h>

// Problem constants (match reference).
#define B_   4096
#define T_   64
#define IN_  2
#define EMB_ 128
#define H_   256
#define G4_  1024  // 4*H

#define MT_  32    // batch rows per workgroup (2 WMMA M-tiles)

typedef __attribute__((ext_vector_type(16))) __bf16 v16bf;
typedef __attribute__((ext_vector_type(8)))  float  v8f;

// ---------------------------------------------------------------------------
// Pack an f32 weight matrix W[K][N] (row-major) into bf16 WMMA B-fragments.
// Fragment order: frag(nt, kt) of 512 bf16 (1 KB). Within a fragment, lane l
// holds 16 contiguous bf16 = W[kt*32 + (j>>3)*16 + (l>>4)*8 + (j&7)][nt*16 + (l&15)], j=0..15:
// the 16-bit B-matrix (32x16) lane layout for v_wmma_f32_16x16x32_bf16.
// Main kernel then does fully-coalesced 32B/lane loads straight from L2.
// ---------------------------------------------------------------------------
__global__ void k_pack_weights(const float* __restrict__ src,
                               __bf16* __restrict__ dst, int K, int N) {
  const int KT = K >> 5;
  const int total = K * N;  // == NT*KT*512
  int e = blockIdx.x * blockDim.x + threadIdx.x;
  if (e >= total) return;
  int frag = e >> 9;
  int lane = (e >> 4) & 31;
  int j    = e & 15;
  int kt   = frag % KT;
  int nt   = frag / KT;
  int k = kt * 32 + (j >> 3) * 16 + (lane >> 4) * 8 + (j & 7);
  int n = nt * 16 + (lane & 15);
  dst[e] = (__bf16)src[(size_t)k * N + n];
}

// ---------------------------------------------------------------------------
// Fused embedding + LSTM. One workgroup (256 thr = 8 waves) per 32-row batch
// tile (two WMMA M-tiles); loops over all T=64 steps. Wave w owns h columns
// [w*32, w*32+32) and the 4 matching gate column-tile pairs, so the gate
// nonlinearity / cell update is wave-local. Each B-fragment load feeds two
// WMMAs (both M-tiles) to halve L2 weight traffic.
// ---------------------------------------------------------------------------
__launch_bounds__(256, 1)
__global__ void k_traj_lstm(const float* __restrict__ traj,
                            const float* __restrict__ Wemb,
                            const float* __restrict__ bemb,
                            const float* __restrict__ blstm,
                            const __bf16* __restrict__ fWx,
                            const __bf16* __restrict__ fWh,
                            float* __restrict__ out) {
  constexpr int SHS = 264;  // sH row stride (bf16): 528B -> 16 rows span all 64 banks
  constexpr int SXS = 136;  // sX row stride (bf16): 272B, same property
  __shared__ __align__(16) __bf16 sH[2][MT_][SHS];  // h_{t-1}/h_t, double buffered
  __shared__ __align__(16) __bf16 sX[MT_][SXS];     // relu(traj @ Wemb + b), step t

  const int tid    = threadIdx.x;
  const int lane   = tid & 31;
  const int wid    = tid >> 5;   // wave 0..7 -> h columns wid*32..wid*32+31
  const int half   = lane >> 4;  // lane group (K/M split per ISA layout)
  const int lcol   = lane & 15;
  const int batch0 = blockIdx.x * MT_;

  // Zero initial hidden state (t=0 reads sH[0]).
  for (int i = tid; i < MT_ * SHS; i += 256)
    ((__bf16*)sH[0])[i] = (__bf16)0.0f;

  // Per-lane biases for this wave's gate columns (constant over t and rows).
  float bI[2], bF[2], bG[2], bO[2];
#pragma unroll
  for (int s = 0; s < 2; ++s) {
    int nh = wid * 32 + s * 16 + lcol;
    bI[s] = blstm[nh];
    bF[s] = blstm[256 + nh];
    bG[s] = blstm[512 + nh];
    bO[s] = blstm[768 + nh];
  }

  float cst[2][2][8];  // cell state  [mtile][subtile][row-in-frag]
  float hst[2][2][8];  // last h (f32) for the hT output
#pragma unroll
  for (int mt = 0; mt < 2; ++mt)
#pragma unroll
    for (int s = 0; s < 2; ++s)
#pragma unroll
      for (int e = 0; e < 8; ++e) { cst[mt][s][e] = 0.0f; hst[mt][s][e] = 0.0f; }

  // Embedding staging: thread -> (row, 16 contiguous cols). 32*128 / 256 = 16.
  const int erow  = tid >> 3;         // 0..31
  const int ecol0 = (tid & 7) * 16;   // 0..112

  int cur = 0;
  for (int t = 0; t < T_; ++t) {
    __syncthreads();  // prev step done reading sX before we overwrite it
    {
      float t0 = traj[((size_t)(batch0 + erow) * T_ + t) * IN_ + 0];
      float t1 = traj[((size_t)(batch0 + erow) * T_ + t) * IN_ + 1];
#pragma unroll
      for (int i = 0; i < 16; ++i) {
        int col = ecol0 + i;
        float v = fmaf(t0, Wemb[col], fmaf(t1, Wemb[EMB_ + col], bemb[col]));
        sX[erow][col] = (__bf16)fmaxf(v, 0.0f);
      }
    }
    __syncthreads();  // sX ready; sH[cur] safe to read

    // Opaque zero element-offset, re-defined every iteration: blocks LICM from
    // hoisting the loop-invariant fragment loads (which previously spilled the
    // whole weight set to scratch), while keeping the *base pointers* intact so
    // address-space inference still selects GLOBAL (not FLAT) loads.
    size_t zoff = 0;
    asm volatile("" : "+s"(zoff));
    const __bf16* wxp = fWx + zoff;
    const __bf16* whp = fWh + zoff;

    v8f acc[2][4][2];  // [mtile][gate i/f/g/o][subtile]
    {
      v8f z = {};
#pragma unroll
      for (int mt = 0; mt < 2; ++mt)
#pragma unroll
        for (int g = 0; g < 4; ++g)
#pragma unroll
          for (int s = 0; s < 2; ++s) acc[mt][g][s] = z;
    }

    // ---- z += x_t @ Wx : K = 128 = 4 k-steps of 32 ----
#pragma unroll
    for (int kt = 0; kt < 4; ++kt) {
      union { v16bf v; uint4 q[2]; } a0, a1;
      const __bf16* r0 = &sX[lcol][kt * 32 + half * 8];
      const __bf16* r1 = &sX[16 + lcol][kt * 32 + half * 8];
      a0.q[0] = *(const uint4*)(r0);       // K = kb + half*8 .. +7
      a0.q[1] = *(const uint4*)(r0 + 16);  // K = kb + 16 + half*8 .. +7
      a1.q[0] = *(const uint4*)(r1);
      a1.q[1] = *(const uint4*)(r1 + 16);
#pragma unroll
      for (int g = 0; g < 4; ++g)
#pragma unroll
        for (int s = 0; s < 2; ++s) {
          int nt = g * 16 + wid * 2 + s;
          v16bf b = *(const v16bf*)(wxp + (((size_t)nt * 4 + kt) << 9) + lane * 16);
          acc[0][g][s] = __builtin_amdgcn_wmma_f32_16x16x32_bf16(
              false, a0.v, false, b, (short)0, acc[0][g][s], false, false);
          acc[1][g][s] = __builtin_amdgcn_wmma_f32_16x16x32_bf16(
              false, a1.v, false, b, (short)0, acc[1][g][s], false, false);
        }
    }
    // ---- z += h_{t-1} @ Wh : K = 256 = 8 k-steps of 32 ----
#pragma unroll
    for (int kt = 0; kt < 8; ++kt) {
      union { v16bf v; uint4 q[2]; } a0, a1;
      const __bf16* r0 = &sH[cur][lcol][kt * 32 + half * 8];
      const __bf16* r1 = &sH[cur][16 + lcol][kt * 32 + half * 8];
      a0.q[0] = *(const uint4*)(r0);
      a0.q[1] = *(const uint4*)(r0 + 16);
      a1.q[0] = *(const uint4*)(r1);
      a1.q[1] = *(const uint4*)(r1 + 16);
#pragma unroll
      for (int g = 0; g < 4; ++g)
#pragma unroll
        for (int s = 0; s < 2; ++s) {
          int nt = g * 16 + wid * 2 + s;
          v16bf b = *(const v16bf*)(whp + (((size_t)nt * 8 + kt) << 9) + lane * 16);
          acc[0][g][s] = __builtin_amdgcn_wmma_f32_16x16x32_bf16(
              false, a0.v, false, b, (short)0, acc[0][g][s], false, false);
          acc[1][g][s] = __builtin_amdgcn_wmma_f32_16x16x32_bf16(
              false, a1.v, false, b, (short)0, acc[1][g][s], false, false);
        }
    }

    // ---- gate combine (wave-local): i,f,g,o -> c, h ----
    const int nxt = cur ^ 1;
#pragma unroll
    for (int mt = 0; mt < 2; ++mt)
#pragma unroll
      for (int s = 0; s < 2; ++s) {
        int nh = wid * 32 + s * 16 + lcol;
#pragma unroll
        for (int e = 0; e < 8; ++e) {
          int m = mt * 16 + half * 8 + e;  // row within the 32-row batch tile
          float iv = 1.0f / (1.0f + __expf(-(acc[mt][0][s][e] + bI[s])));
          float fv = 1.0f / (1.0f + __expf(-(acc[mt][1][s][e] + bF[s])));
          float gv = tanhf(acc[mt][2][s][e] + bG[s]);
          float ov = 1.0f / (1.0f + __expf(-(acc[mt][3][s][e] + bO[s])));
          float c = fv * cst[mt][s][e] + iv * gv;
          cst[mt][s][e] = c;
          float h = ov * tanhf(c);
          hst[mt][s][e] = h;
          sH[nxt][m][nh] = (__bf16)h;  // next step's recurrence input (bf16)
          out[((size_t)(batch0 + m) * T_ + t) * H_ + nh] = h;  // hs[b][t][h]
        }
      }
    cur = nxt;
  }

  // Final hT and cT (concatenated after hs in return order).
  const size_t OFF1 = (size_t)B_ * T_ * H_;
  const size_t OFF2 = OFF1 + (size_t)B_ * H_;
#pragma unroll
  for (int mt = 0; mt < 2; ++mt)
#pragma unroll
    for (int s = 0; s < 2; ++s) {
      int nh = wid * 32 + s * 16 + lcol;
#pragma unroll
      for (int e = 0; e < 8; ++e) {
        int m = mt * 16 + half * 8 + e;
        out[OFF1 + (size_t)(batch0 + m) * H_ + nh] = hst[mt][s][e];
        out[OFF2 + (size_t)(batch0 + m) * H_ + nh] = cst[mt][s][e];
      }
    }
}

// ---------------------------------------------------------------------------
extern "C" void kernel_launch(void* const* d_in, const int* in_sizes, int n_in,
                              void* d_out, int out_size, void* d_ws, size_t ws_size,
                              hipStream_t stream) {
  (void)in_sizes; (void)n_in; (void)out_size; (void)ws_size;
  const float* traj  = (const float*)d_in[0];  // [B, T, IN]
  const float* Wemb  = (const float*)d_in[1];  // [IN, EMB]
  const float* bemb  = (const float*)d_in[2];  // [EMB]
  const float* Wx    = (const float*)d_in[3];  // [EMB, 4H]
  const float* Wh    = (const float*)d_in[4];  // [H, 4H]
  const float* blstm = (const float*)d_in[5];  // [4H]
  float* out = (float*)d_out;

  __bf16* fWx = (__bf16*)d_ws;                  // 128*1024 bf16 = 256 KB
  __bf16* fWh = fWx + (size_t)EMB_ * G4_;       // 256*1024 bf16 = 512 KB

  // Pre-swizzle weights into WMMA B-fragment order (bf16). Re-done every call:
  // deterministic, and keeps weights L2-resident for the main kernel.
  k_pack_weights<<<(EMB_ * G4_ + 255) / 256, 256, 0, stream>>>(Wx, fWx, EMB_, G4_);
  k_pack_weights<<<(H_   * G4_ + 255) / 256, 256, 0, stream>>>(Wh, fWh, H_,   G4_);

  // 128 workgroups x 8 waves; each owns a 32-row batch tile across all 64 steps.
  k_traj_lstm<<<B_ / MT_, 256, 0, stream>>>(traj, Wemb, bemb, blstm, fWx, fWh, out);
}
